// BiLSTM_CRF_73194832658536
// MI455X (gfx1250) — compile-verified, matrix operand order checked by reading, not run on hardware
//
#include <hip/hip_runtime.h>
#include <hip/hip_bf16.h>

// ---------------------------------------------------------------------------
// BiLSTM-CRF forward for MI455X (gfx1250, wave32, WMMA).
//   V=50000 D=256 H2=256 (per dir) K=64 B=128 T=512
// Plan:
//   1. pack W_ih_f/b, W_hh_f/b, W_tag into WMMA B-fragment layout (f16)
//   2. gather embeddings -> X f16 [T*B, 256]
//   3. Gx_d = X @ W_ih_d.T + b_d  (WMMA GEMM, f16 out, bias baked)
//   4. persistent recurrent kernel: 512 steps, grid-wide barrier, both dirs
//   5. emissions = (h_all @ W_tag.T + b_tag) * mask  (WMMA GEMM)
//   6. CRF forward recursion, one 64-thread block per batch row
// ---------------------------------------------------------------------------

typedef _Float16 v16h __attribute__((ext_vector_type(16)));
typedef _Float16 v8h  __attribute__((ext_vector_type(8)));
typedef float    v8f  __attribute__((ext_vector_type(8)));

#define T_LEN 512
#define B_SZ  128
#define D_IN  256
#define H2_SZ 256
#define NG    1024          // 4*H2
#define K_TAG 64
#define NEGV  (-10000.0f)

// ---- WMMA fragment helpers -------------------------------------------------
// A (16x32 f16, MxK) from row-major f16 with leading dim `ld`:
// lane<16: M=lane, K 0..7 then 16..23 ; lane>=16: M=lane-16, K 8..15 then 24..31
static __device__ __forceinline__ v16h load_a_frag(const _Float16* base, int ld) {
  const int lane = threadIdx.x & 31;
  const int m = lane & 15, s = lane >> 4;
  const _Float16* p = base + (size_t)m * ld + s * 8;
  v8h lo = *(const v8h*)(p);
  v8h hi = *(const v8h*)(p + 16);
  return __builtin_shufflevector(lo, hi, 0,1,2,3,4,5,6,7,8,9,10,11,12,13,14,15);
}

// B (32x16 f16, KxN) from pre-packed per-tile buffer: 512 halfs/tile, lane*16
static __device__ __forceinline__ v16h load_b_frag(const _Float16* tile) {
  const int lane = threadIdx.x & 31;
  return *(const v16h*)(tile + lane * 16);
}

static __device__ __forceinline__ float sigf(float x) {
  return 1.0f / (1.0f + __expf(-x));
}

// ---- 1. pack W [N][K] row-major f32 -> B-fragment f16 buffer ----------------
// element (k_local, n_local) of tile (kt, nt): lane = n_local + 16*(k>>3 & 1),
// e = (k&7) + 8*(k>=16).  dst[tile*512 + lane*16 + e]
__global__ void pack_b_frag_kernel(const float* __restrict__ W,
                                   _Float16* __restrict__ dst, int N, int K) {
  int tid = blockIdx.x * 256 + threadIdx.x;
  if (tid >= N * K) return;
  int tile = tid >> 9;
  int r = tid & 511;
  int lane = r >> 4, e = r & 15;
  int ntn = N >> 4;
  int kt = tile / ntn, nt = tile - kt * ntn;
  int s = lane >> 4;
  int k = kt * 32 + s * 8 + (e & 7) + ((e >= 8) ? 16 : 0);
  int n = nt * 16 + (lane & 15);
  dst[tid] = (_Float16)W[(size_t)n * K + k];
}

// ---- 2. embedding gather, f32 -> f16, layout [T*B, D] (m = t*128 + b) -------
__global__ void embed_kernel(const int* __restrict__ inp,
                             const float* __restrict__ emb,
                             _Float16* __restrict__ X) {
  int m = blockIdx.x;
  int t = m >> 7, b = m & 127;
  int tok = inp[b * T_LEN + t];
  X[(size_t)m * D_IN + threadIdx.x] = (_Float16)emb[(size_t)tok * D_IN + threadIdx.x];
}

// ---- 3. Gx = X @ W_ih.T + bias : [65536 x 1024], K=256 ----------------------
// 512 blocks x 8 waves; each wave owns one 16-row M-tile, sweeps 64 N-tiles.
__global__ void __launch_bounds__(256) gx_gemm_kernel(
    const _Float16* __restrict__ X, const _Float16* __restrict__ wfrag,
    const float* __restrict__ bias, _Float16* __restrict__ Gx) {
  const int wave = threadIdx.x >> 5;
  const int lane = threadIdx.x & 31;
  const int s = lane >> 4, nl = lane & 15;
  const int row0 = (blockIdx.x * 8 + wave) * 16;

  v16h a[8];
#pragma unroll
  for (int kt = 0; kt < 8; ++kt)
    a[kt] = load_a_frag(X + (size_t)row0 * D_IN + kt * 32, D_IN);

  for (int nt = 0; nt < 64; ++nt) {
    const int n = nt * 16 + nl;
    const float bv = bias[n];
    v8f acc;
#pragma unroll
    for (int r = 0; r < 8; ++r) acc[r] = bv;
#pragma unroll
    for (int kt = 0; kt < 8; ++kt) {
      v16h bfr = load_b_frag(wfrag + (size_t)(kt * 64 + nt) * 512);
      acc = __builtin_amdgcn_wmma_f32_16x16x32_f16(false, a[kt], false, bfr,
                                                   (short)0, acc, false, false);
    }
#pragma unroll
    for (int r = 0; r < 8; ++r)
      Gx[(size_t)(row0 + s * 8 + r) * NG + n] = (_Float16)acc[r];
  }
}

// ---- device-wide barrier (monotonic counter) --------------------------------
static __device__ __forceinline__ void grid_sync(unsigned* cnt, unsigned target) {
  __threadfence();
  __syncthreads();
  if (threadIdx.x == 0) {
    atomicAdd(cnt, 1u);
    while (atomicAdd(cnt, 0u) < target) __builtin_amdgcn_s_sleep(2);
  }
  __syncthreads();
  __threadfence();
}

// ---- 4. persistent recurrent LSTM (both directions), 32 WGs -----------------
// blockIdx: dir = bx>>4, j = bx&15 -> WG owns hidden cols [j*16, j*16+16) for
// all 4 gates; 8 waves = 8 M-tiles covering B=128.  c state in VGPRs.
__global__ void __launch_bounds__(256) lstm_recurrent_kernel(
    const _Float16* __restrict__ Gx_f, const _Float16* __restrict__ Gx_b,
    const _Float16* __restrict__ whhfrag_f, const _Float16* __restrict__ whhfrag_b,
    _Float16* __restrict__ hcur,   // [2 dir][2 buf][128][256] f16, zero-init
    _Float16* __restrict__ h_all,  // [T*B][512] f16
    unsigned* __restrict__ cnt) {
  const int dir  = blockIdx.x >> 4;
  const int j    = blockIdx.x & 15;
  const int wave = threadIdx.x >> 5;
  const int lane = threadIdx.x & 31;
  const int s = lane >> 4, nl = lane & 15;
  const int row0 = wave * 16;
  const int col0 = j * 16;

  const _Float16* Gx    = dir ? Gx_b : Gx_f;
  const _Float16* wfrag = dir ? whhfrag_b : whhfrag_f;
  _Float16* hbase = hcur + (size_t)dir * 2 * B_SZ * H2_SZ;

  // W_hh slice for this WG: 4 gates x 8 k-tiles = 32 tiles = 32 KB in LDS
  __shared__ _Float16 s_b[4 * 8 * 512];
  for (int idx = threadIdx.x; idx < 4 * 8 * 512; idx += 256) {
    int tile = idx >> 9;
    int g = tile >> 3, kt = tile & 7;
    int gtile = kt * 64 + (g * 16 + j);          // N/16 = 64 n-tiles per k-tile
    s_b[idx] = wfrag[(size_t)gtile * 512 + (idx & 511)];
  }
  __syncthreads();

  v8f creg;
#pragma unroll
  for (int r = 0; r < 8; ++r) creg[r] = 0.0f;

  for (int t = 0; t < T_LEN; ++t) {
    const int tcur = dir ? (T_LEN - 1 - t) : t;
    const _Float16* rb = hbase + (size_t)(t & 1) * B_SZ * H2_SZ;
    _Float16*       wb = hbase + (size_t)((t + 1) & 1) * B_SZ * H2_SZ;

    v16h a[8];
#pragma unroll
    for (int kt = 0; kt < 8; ++kt)
      a[kt] = load_a_frag(rb + (size_t)row0 * H2_SZ + kt * 32, H2_SZ);

    const _Float16* gx = Gx + (size_t)(tcur * B_SZ + row0) * NG;
    v8f gate[4];
#pragma unroll
    for (int g = 0; g < 4; ++g) {
      const int n = g * H2_SZ + col0 + nl;
      v8f acc;
#pragma unroll
      for (int r = 0; r < 8; ++r) acc[r] = (float)gx[(size_t)(s * 8 + r) * NG + n];
#pragma unroll
      for (int kt = 0; kt < 8; ++kt) {
        v16h bfr = *(const v16h*)(s_b + ((g * 8 + kt) << 9) + lane * 16);
        acc = __builtin_amdgcn_wmma_f32_16x16x32_f16(false, a[kt], false, bfr,
                                                     (short)0, acc, false, false);
      }
      gate[g] = acc;
    }

#pragma unroll
    for (int r = 0; r < 8; ++r) {
      float iv = sigf(gate[0][r]);
      float fv = sigf(gate[1][r]);
      float gv = tanhf(gate[2][r]);
      float ov = sigf(gate[3][r]);
      creg[r] = fv * creg[r] + iv * gv;
      float hv = ov * tanhf(creg[r]);
      int row = row0 + s * 8 + r;
      int col = col0 + nl;
      wb[(size_t)row * H2_SZ + col] = (_Float16)hv;
      h_all[(size_t)(tcur * B_SZ + row) * (2 * H2_SZ) + dir * H2_SZ + col] =
          (_Float16)hv;
    }
    grid_sync(cnt, 32u * (unsigned)(t + 1));
  }
}

// ---- 5. emissions = (h_all @ W_tag.T + b_tag) * mask : [65536 x 64], K=512 --
__global__ void __launch_bounds__(256) emis_kernel(
    const _Float16* __restrict__ h_all, const _Float16* __restrict__ wtagfrag,
    const float* __restrict__ b_tag, const int* __restrict__ inp,
    float* __restrict__ emis) {
  const int wave = threadIdx.x >> 5;
  const int lane = threadIdx.x & 31;
  const int s = lane >> 4, nl = lane & 15;
  const int row0 = (blockIdx.x * 8 + wave) * 16;

  v8f acc[4];
#pragma unroll
  for (int nt = 0; nt < 4; ++nt) {
    float bv = b_tag[nt * 16 + nl];
#pragma unroll
    for (int r = 0; r < 8; ++r) acc[nt][r] = bv;
  }
  for (int kt = 0; kt < 16; ++kt) {
    v16h a = load_a_frag(h_all + (size_t)row0 * 512 + kt * 32, 512);
#pragma unroll
    for (int nt = 0; nt < 4; ++nt) {
      v16h bfr = load_b_frag(wtagfrag + (size_t)(kt * 4 + nt) * 512);
      acc[nt] = __builtin_amdgcn_wmma_f32_16x16x32_f16(false, a, false, bfr,
                                                       (short)0, acc[nt], false, false);
    }
  }
#pragma unroll
  for (int r = 0; r < 8; ++r) {
    int row = row0 + s * 8 + r;
    int t = row >> 7, b = row & 127;
    float mv = (inp[b * T_LEN + t] != 2) ? 1.0f : 0.0f;
#pragma unroll
    for (int nt = 0; nt < 4; ++nt)
      emis[(size_t)row * K_TAG + nt * 16 + nl] = acc[nt][r] * mv;
  }
}

// ---- 6. CRF forward recursion: one block per batch row ----------------------
__global__ void __launch_bounds__(64) crf_kernel(
    const float* __restrict__ emis,   // [T*B][64], row = t*128 + b
    const float* __restrict__ trans,  // [64][64]
    const int* __restrict__ inp, float* __restrict__ out) {
  __shared__ float s_trans[K_TAG * K_TAG];
  __shared__ float s_score[K_TAG];
  __shared__ float s_fin[K_TAG];
  const int b = blockIdx.x, i = threadIdx.x;

  for (int idx = i; idx < K_TAG * K_TAG; idx += 64) s_trans[idx] = trans[idx];
  s_score[i] = (i == 1) ? 0.0f : NEGV;   // STOP_ID = 1
  __syncthreads();

  const float* tr = s_trans + i * K_TAG;
  for (int t = 0; t < T_LEN; ++t) {
    float emit = emis[(size_t)(t * B_SZ + b) * K_TAG + i];
    float mt = (inp[b * T_LEN + t] != 2) ? 1.0f : 0.0f;
    float mx = -3.4e38f;
    for (int jj = 0; jj < K_TAG; ++jj) mx = fmaxf(mx, s_score[jj] + tr[jj]);
    float sum = 0.0f;
    for (int jj = 0; jj < K_TAG; ++jj) sum += __expf(s_score[jj] + tr[jj] - mx);
    float nv = mx + __logf(sum) + emit;
    float res = nv * mt + s_score[i] * (1.0f - mt);
    __syncthreads();
    s_score[i] = res;
    __syncthreads();
  }
  s_fin[i] = s_score[i] + s_trans[1 * K_TAG + i];   // + trans[STOP_ID]
  __syncthreads();
  if (i == 0) {
    float mx = -3.4e38f;
    for (int jj = 0; jj < K_TAG; ++jj) mx = fmaxf(mx, s_fin[jj]);
    float sum = 0.0f;
    for (int jj = 0; jj < K_TAG; ++jj) sum += __expf(s_fin[jj] - mx);
    out[b] = mx + __logf(sum);
  }
}

// ---------------------------------------------------------------------------
extern "C" void kernel_launch(void* const* d_in, const int* in_sizes, int n_in,
                              void* d_out, int out_size, void* d_ws, size_t ws_size,
                              hipStream_t stream) {
  (void)in_sizes; (void)n_in; (void)out_size; (void)ws_size;
  const int*   inp    = (const int*)  d_in[0];
  const float* emb    = (const float*)d_in[1];
  const float* W_ih_f = (const float*)d_in[2];
  const float* W_hh_f = (const float*)d_in[3];
  const float* b_f    = (const float*)d_in[4];
  const float* W_ih_b = (const float*)d_in[5];
  const float* W_hh_b = (const float*)d_in[6];
  const float* b_b    = (const float*)d_in[7];
  const float* W_tag  = (const float*)d_in[8];
  const float* b_tag  = (const float*)d_in[9];
  const float* trans  = (const float*)d_in[10];
  float* out = (float*)d_out;

  char* ws = (char*)d_ws;
  auto alloc = [&](size_t bytes) {
    char* p = ws;
    ws += (bytes + 255) & ~(size_t)255;
    return p;
  };
  const size_t M = (size_t)T_LEN * B_SZ;               // 65536 rows
  _Float16* X     = (_Float16*)alloc(M * D_IN * 2);    //  32 MB
  _Float16* Gx_f  = (_Float16*)alloc(M * NG * 2);      // 128 MB
  _Float16* Gx_b  = (_Float16*)alloc(M * NG * 2);      // 128 MB
  _Float16* h_all = (_Float16*)alloc(M * 512 * 2);     //  64 MB
  float*    emis  = (float*)   alloc(M * K_TAG * 4);   //  16 MB
  _Float16* f_ihf = (_Float16*)alloc((size_t)NG * D_IN * 2);
  _Float16* f_ihb = (_Float16*)alloc((size_t)NG * D_IN * 2);
  _Float16* f_hhf = (_Float16*)alloc((size_t)NG * H2_SZ * 2);
  _Float16* f_hhb = (_Float16*)alloc((size_t)NG * H2_SZ * 2);
  _Float16* f_tag = (_Float16*)alloc((size_t)K_TAG * 512 * 2);
  _Float16* hcur  = (_Float16*)alloc((size_t)2 * 2 * B_SZ * H2_SZ * 2);
  unsigned* cnt   = (unsigned*)alloc(256);

  hipMemsetAsync(cnt, 0, 256, stream);
  hipMemsetAsync(hcur, 0, (size_t)2 * 2 * B_SZ * H2_SZ * 2, stream);

  // 1. pack weights into WMMA B-fragment layout
  pack_b_frag_kernel<<<(NG * D_IN) / 256, 256, 0, stream>>>(W_ih_f, f_ihf, NG, D_IN);
  pack_b_frag_kernel<<<(NG * D_IN) / 256, 256, 0, stream>>>(W_ih_b, f_ihb, NG, D_IN);
  pack_b_frag_kernel<<<(NG * H2_SZ) / 256, 256, 0, stream>>>(W_hh_f, f_hhf, NG, H2_SZ);
  pack_b_frag_kernel<<<(NG * H2_SZ) / 256, 256, 0, stream>>>(W_hh_b, f_hhb, NG, H2_SZ);
  pack_b_frag_kernel<<<(K_TAG * 512) / 256, 256, 0, stream>>>(W_tag, f_tag, K_TAG, 512);

  // 2. embedding gather + cast
  embed_kernel<<<(int)M, 256, 0, stream>>>(inp, emb, X);

  // 3. input projections (bias baked in)
  gx_gemm_kernel<<<512, 256, 0, stream>>>(X, f_ihf, b_f, Gx_f);
  gx_gemm_kernel<<<512, 256, 0, stream>>>(X, f_ihb, b_b, Gx_b);

  // 4. recurrence, both directions in one persistent launch
  lstm_recurrent_kernel<<<32, 256, 0, stream>>>(Gx_f, Gx_b, f_hhf, f_hhb,
                                                hcur, h_all, cnt);

  // 5. emissions
  emis_kernel<<<512, 256, 0, stream>>>(h_all, f_tag, b_tag, inp, emis);

  // 6. CRF log-partition
  crf_kernel<<<B_SZ, 64, 0, stream>>>(emis, trans, inp, out);
}
